// MultiHeadAttention_48533130445176
// MI455X (gfx1250) — matrix operand order repaired and checked by reading, not compiled, not run on hardware
//
#include <hip/hip_runtime.h>

#define E_DIM 1024
#define F_DIM 3072
#define NHEAD 16
#define HDIM  64
#define NSEQ  2048
#define BATCH 2
#define MROWS (BATCH * NSEQ)   // 4096
#define BHEAD (BATCH * NHEAD)  // 32

typedef __attribute__((ext_vector_type(16))) _Float16 v16h;
typedef __attribute__((ext_vector_type(8)))  _Float16 v8h;
typedef __attribute__((ext_vector_type(8)))  float    v8f;
typedef unsigned int v4u __attribute__((ext_vector_type(4)));
typedef int          v4i __attribute__((ext_vector_type(4)));
typedef int          v8i __attribute__((ext_vector_type(8)));

union FragU { v16h v; v8h h[2]; };
union H8U  { v8h v; _Float16 e[8]; int i[4]; };

#if defined(__has_builtin)
#  if __has_builtin(__builtin_amdgcn_tensor_load_to_lds)
#    define USE_TDM 1
#  endif
#endif
#ifndef USE_TDM
#  define USE_TDM 0
#endif

// --- WMMA wrapper (D = A x B + C, f16 in, f32 acc) ------------------------
__device__ __forceinline__ v8f wmma_f16(v16h a, v16h b, v8f c) {
  return __builtin_amdgcn_wmma_f32_16x16x32_f16(
      /*neg_a=*/false, a, /*neg_b=*/false, b,
      /*c_mod=*/(short)0, c, /*reuse_a=*/false, /*reuse_b=*/false);
}

// --- A-fragment: 16x32 (MxK), row-major source, ld in elements ------------
// Contiguous k order: lane m<16 -> K 0..15; lane m+16 -> K 16..31
__device__ __forceinline__ v16h frag_a_load(const _Float16* tile, int ld, int lane) {
  const int m  = lane & 15;
  const int hp = lane >> 4;
  const _Float16* r = tile + (size_t)m * ld + hp * 16;
  FragU f;
  f.h[0] = *(const v8h*)(r);
  f.h[1] = *(const v8h*)(r + 8);
  return f.v;
}

// --- B-fragment: 32x16 (KxN); column n = row n of row-major src (A x src^T)
// lane n<16 -> K 0..15; lane n+16 -> K 16..31
__device__ __forceinline__ v16h frag_b_load(const _Float16* tile, int ld, int lane) {
  const int n  = lane & 15;
  const int hp = lane >> 4;
  const _Float16* r = tile + (size_t)n * ld + hp * 16;
  FragU f;
  f.h[0] = *(const v8h*)(r);
  f.h[1] = *(const v8h*)(r + 8);
  return f.v;
}

// --- B-fragment from an LDS tile with ld=32 halves ------------------------
__device__ __forceinline__ v16h frag_b_lds(const _Float16* tile, int row0, int lane) {
  const int n  = lane & 15;
  const int hp = lane >> 4;
  const _Float16* r = tile + (row0 + n) * 32 + hp * 16;
  FragU f;
  f.h[0] = *(const v8h*)(r);
  f.h[1] = *(const v8h*)(r + 8);
  return f.v;
}

#if USE_TDM
// --- Tensor Data Mover: 2-D tile (f16) global -> LDS ----------------------
// D# per CDNA5 ISA 8.3/8.4: group0 = {flags, lds_addr, global_addr, type=2},
// group1 = {data_size=2B, tensor dims, tile dims, dim0 stride}.
// clang-23 builtin arity: (v4u g0, v8i g1, v4i g2, v4i g3, v8i g4, i32 cpol)
__device__ __forceinline__ void tdm_load_tile_f16(
    unsigned lds_off, const _Float16* gptr,
    unsigned tile_d0, unsigned tile_d1,
    unsigned tensor_d0, unsigned tensor_d1, unsigned stride0) {
  unsigned long long ga = (unsigned long long)(uintptr_t)gptr;
  v4u g0;
  g0[0] = 1u;                                        // count=1 (valid user D#)
  g0[1] = lds_off;                                   // lds_addr (bytes)
  g0[2] = (unsigned)(ga & 0xffffffffu);              // global_addr[31:0]
  g0[3] = (unsigned)((ga >> 32) & 0x01ffffffu)       // global_addr[56:32]
          | (2u << 30);                              // type = 2 ("image")
  v8i g1;
  g1[0] = (int)(1u << 16);                           // data_size = 1 -> 2 bytes
  g1[1] = (int)(tensor_d0 << 16);                    // tensor_dim0[15:0]
  g1[2] = (int)((tensor_d0 >> 16) | (tensor_d1 << 16));
  g1[3] = (int)((tensor_d1 >> 16) | (tile_d0 << 16));// tile_dim0
  g1[4] = (int)(tile_d1 & 0xffffu);                  // tile_dim1 (tile_dim2=0)
  g1[5] = (int)stride0;                              // tensor_dim0_stride[31:0]
  g1[6] = 0;
  g1[7] = 0;
  v4i z4 = {0, 0, 0, 0};
  v8i z8 = {0, 0, 0, 0, 0, 0, 0, 0};
  __builtin_amdgcn_tensor_load_to_lds(g0, g1, z4, z4, z8, 0);
}
#endif

// --- f32 -> f16 conversion ------------------------------------------------
__global__ void cvt_f32_f16(const float* __restrict__ src,
                            _Float16* __restrict__ dst, int n) {
  int i = blockIdx.x * blockDim.x + threadIdx.x;
  if (i < n) dst[i] = (_Float16)src[i];
}

// --- stage one 128x32 f16 W tile into LDS buffer --------------------------
__device__ __forceinline__ void stage_w_tile(
    _Float16* btile_buf, unsigned lds_off_buf,
    const _Float16* __restrict__ wh, int n_blk, int k0,
    int wave, int tid) {
#if USE_TDM
  (void)btile_buf; (void)tid;
  if (wave == 0) {
    tdm_load_tile_f16(lds_off_buf, wh + (size_t)n_blk * E_DIM + k0,
                      /*tile_d0=*/32, /*tile_d1=*/128,
                      /*tensor_d0=*/E_DIM, /*tensor_d1=*/F_DIM,
                      /*stride0=*/E_DIM);
  }
#else
  (void)lds_off_buf; (void)wave;
  const int row = tid >> 1;            // 0..127
  const int co  = (tid & 1) * 16;      // 0 or 16
  const _Float16* src = wh + (size_t)(n_blk + row) * E_DIM + k0 + co;
  _Float16* dst = btile_buf + row * 32 + co;
  *(v8h*)dst       = *(const v8h*)src;
  *(v8h*)(dst + 8) = *(const v8h*)(src + 8);
#endif
}

// --- QKV projection: qkv[m,f] = sum_e x[m,e]*W[f,e] + b[f] ----------------
// W tile staged to LDS via TDM (double buffered); A direct from global.
// Writes Q (pre-scaled by hd^-0.5) [BH,N,64], K [BH,N,64], V^T [BH,64,N].
__global__ __launch_bounds__(256)
void qkv_gemm(const _Float16* __restrict__ xh, const _Float16* __restrict__ wh,
              const float* __restrict__ bias,
              _Float16* __restrict__ Qs, _Float16* __restrict__ Kc,
              _Float16* __restrict__ Vt) {
  __shared__ __align__(16) _Float16 btile[2][128 * 32];   // 2 x 8KB

  const int tid   = threadIdx.x;
  const int lane  = tid & 31;
  const int wave  = tid >> 5;
  const int m_base = blockIdx.y * 128 + (wave & 3) * 32;   // 4 row-groups
  const int n_blk  = blockIdx.x * 128;
  const int wn     = (wave >> 2) * 64;                     // 2 col-groups

  const unsigned lds0 = (unsigned)(uintptr_t)(&btile[0][0]);

  v8f acc[2][4] = {};

  // prologue: stage first tile into buffer 0
  stage_w_tile(&btile[0][0], lds0, wh, n_blk, 0, wave, tid);

  for (int k0 = 0; k0 < E_DIM; k0 += 32) {
    const int buf = (k0 >> 5) & 1;
#if USE_TDM
    if (wave == 0) __builtin_amdgcn_s_wait_tensorcnt(0);
#endif
    __syncthreads();   // tile(buf) ready; buf^1 fully consumed in iter-1

    if (k0 + 32 < E_DIM)
      stage_w_tile(&btile[buf ^ 1][0], lds0 + (unsigned)(buf ^ 1) * 8192u,
                   wh, n_blk, k0 + 32, wave, tid);

    v16h a0 = frag_a_load(xh + (size_t)m_base * E_DIM + k0, E_DIM, lane);
    v16h a1 = frag_a_load(xh + (size_t)(m_base + 16) * E_DIM + k0, E_DIM, lane);
#pragma unroll
    for (int j = 0; j < 4; ++j) {
      v16h b = frag_b_lds(&btile[buf][0], wn + j * 16, lane);
      acc[0][j] = wmma_f16(a0, b, acc[0][j]);
      acc[1][j] = wmma_f16(a1, b, acc[1][j]);
    }
  }

  // Epilogue: bias add + scatter into Q/K/V^T (f -> head h = f/192, c = f%192)
  const int nl = lane & 15;
  const int hp = lane >> 4;
#pragma unroll
  for (int j = 0; j < 4; ++j) {
    const int col = n_blk + wn + j * 16 + nl;
    const float bv = bias[col];
    const int h = col / (3 * HDIM);
    const int c = col % (3 * HDIM);
#pragma unroll
    for (int i = 0; i < 2; ++i) {
#pragma unroll
      for (int r = 0; r < 8; ++r) {
        const int m = m_base + i * 16 + (hp ? (8 + r) : r);
        const float val = acc[i][j][r] + bv;
        const int bb = m >> 11;            // m / 2048
        const int nn = m & (NSEQ - 1);
        const int bh = bb * NHEAD + h;
        if (c < HDIM) {
          Qs[((size_t)bh * NSEQ + nn) * HDIM + c] = (_Float16)(val * 0.125f);
        } else if (c < 2 * HDIM) {
          Kc[((size_t)bh * NSEQ + nn) * HDIM + (c - HDIM)] = (_Float16)val;
        } else {
          Vt[((size_t)bh * HDIM + (c - 2 * HDIM)) * NSEQ + nn] = (_Float16)val;
        }
      }
    }
  }
}

// --- Flash attention: each wave owns 16 queries; online softmax -----------
// Computes S^T = K·Q^T so softmax state is per-lane; O^T = V^T · P^T.
__global__ __launch_bounds__(256)
void flash_attn(const _Float16* __restrict__ Qs, const _Float16* __restrict__ Kc,
                const _Float16* __restrict__ Vt, float* __restrict__ out) {
  const int lane = threadIdx.x & 31;
  const int wave = threadIdx.x >> 5;
  const int bh   = blockIdx.y;
  const int q0   = blockIdx.x * 128 + wave * 16;

  const _Float16* Qp = Qs + ((size_t)bh * NSEQ + q0) * HDIM;
  const _Float16* Kp = Kc + (size_t)bh * NSEQ * HDIM;
  const _Float16* Vp = Vt + (size_t)bh * HDIM * NSEQ;

  const v16h qb0 = frag_b_load(Qp, HDIM, lane);
  const v16h qb1 = frag_b_load(Qp + 32, HDIM, lane);

  float m_run = -3.0e38f;
  float l_run = 0.0f;
  v8f o[4] = {};

  for (int k0 = 0; k0 < NSEQ; k0 += 32) {
    if (k0 + 32 < NSEQ) {
      __builtin_prefetch(Kp + (size_t)(k0 + 32) * HDIM, 0, 1);
      __builtin_prefetch(Vp + k0 + 32, 0, 1);
    }

    // S^T chunk: s0 = keys k0..k0+15, s1 = keys k0+16..k0+31 (cols = q)
    v8f s0 = {}, s1 = {};
    s0 = wmma_f16(frag_a_load(Kp + (size_t)k0 * HDIM,        HDIM, lane), qb0, s0);
    s0 = wmma_f16(frag_a_load(Kp + (size_t)k0 * HDIM + 32,   HDIM, lane), qb1, s0);
    s1 = wmma_f16(frag_a_load(Kp + (size_t)(k0 + 16) * HDIM,      HDIM, lane), qb0, s1);
    s1 = wmma_f16(frag_a_load(Kp + (size_t)(k0 + 16) * HDIM + 32, HDIM, lane), qb1, s1);

    float mx = s0[0];
#pragma unroll
    for (int r = 1; r < 8; ++r) mx = fmaxf(mx, s0[r]);
#pragma unroll
    for (int r = 0; r < 8; ++r) mx = fmaxf(mx, s1[r]);
    mx = fmaxf(mx, __shfl_xor(mx, 16, 32));

    const float m_new = fmaxf(m_run, mx);
    const float corr  = __expf(m_run - m_new);

    float psum = 0.0f;
    H8U plo, phi;
#pragma unroll
    for (int r = 0; r < 8; ++r) {
      const float e0 = __expf(s0[r] - m_new);
      const float e1 = __expf(s1[r] - m_new);
      psum += e0 + e1;
      plo.e[r] = (_Float16)e0;
      phi.e[r] = (_Float16)e1;
    }
    psum += __shfl_xor(psum, 16, 32);
    l_run = l_run * corr + psum;
    m_run = m_new;
#pragma unroll
    for (int t = 0; t < 4; ++t)
#pragma unroll
      for (int r = 0; r < 8; ++r) o[t][r] *= corr;

    // P^T C-frag -> B-frag: exchange missing 8 keys with xor-16 partner lane
    H8U plo_x, phi_x;
#pragma unroll
    for (int w = 0; w < 4; ++w) {
      plo_x.i[w] = __shfl_xor(plo.i[w], 16, 32);
      phi_x.i[w] = __shfl_xor(phi.i[w], 16, 32);
    }
    FragU pf;
    pf.h[0] = (lane < 16) ? plo.v : phi_x.v;  // keys 0-7  | 16-23
    pf.h[1] = (lane < 16) ? plo_x.v : phi.v;  // keys 8-15 | 24-31

    // O^T += V^T · P^T  (A = V^T tile [16d x 32k], B = P^T)
#pragma unroll
    for (int t = 0; t < 4; ++t) {
      v16h va = frag_a_load(Vp + (size_t)(t * 16) * NSEQ + k0, NSEQ, lane);
      o[t] = wmma_f16(va, pf.v, o[t]);
    }
  }

  const float linv = 1.0f / l_run;
  const int nl = lane & 15;
  const int hp = lane >> 4;
  const int b  = bh / NHEAD;
  const int h  = bh % NHEAD;
  const int q  = q0 + nl;
  float* orow = out + ((size_t)b * NSEQ + q) * E_DIM + h * HDIM;
#pragma unroll
  for (int t = 0; t < 4; ++t)
#pragma unroll
    for (int r = 0; r < 8; ++r)
      orow[t * 16 + (hp ? (8 + r) : r)] = o[t][r] * linv;
}

// --------------------------------------------------------------------------
extern "C" void kernel_launch(void* const* d_in, const int* in_sizes, int n_in,
                              void* d_out, int out_size, void* d_ws, size_t ws_size,
                              hipStream_t stream) {
  const float* x    = (const float*)d_in[0];   // [2, 2048, 1024]
  const float* W    = (const float*)d_in[1];   // [3072, 1024]
  const float* bias = (const float*)d_in[2];   // [3072]
  float* out = (float*)d_out;                  // [2, 2048, 1024]

  char* ws = (char*)d_ws;
  size_t off = 0;
  _Float16* xh = (_Float16*)(ws + off); off += (size_t)MROWS * E_DIM * sizeof(_Float16);
  _Float16* wh = (_Float16*)(ws + off); off += (size_t)F_DIM * E_DIM * sizeof(_Float16);
  _Float16* Qs = (_Float16*)(ws + off); off += (size_t)BHEAD * NSEQ * HDIM * sizeof(_Float16);
  _Float16* Kc = (_Float16*)(ws + off); off += (size_t)BHEAD * NSEQ * HDIM * sizeof(_Float16);
  _Float16* Vt = (_Float16*)(ws + off); off += (size_t)BHEAD * HDIM * NSEQ * sizeof(_Float16);
  (void)ws_size; (void)in_sizes; (void)n_in; (void)out_size;

  const int nx = MROWS * E_DIM;   // 4,194,304
  const int nw = F_DIM * E_DIM;   // 3,145,728
  cvt_f32_f16<<<(nx + 255) / 256, 256, 0, stream>>>(x, xh, nx);
  cvt_f32_f16<<<(nw + 255) / 256, 256, 0, stream>>>(W, wh, nw);

  qkv_gemm<<<dim3(F_DIM / 128, MROWS / 128), 256, 0, stream>>>(xh, wh, bias, Qs, Kc, Vt);
  flash_attn<<<dim3(NSEQ / 128, BHEAD), 256, 0, stream>>>(Qs, Kc, Vt, out);
}
